// FGIB_59545426592269
// MI455X (gfx1250) — compile-verified, matrix operand order checked by reading, not run on hardware
//
#include <hip/hip_runtime.h>
#include <hip/hip_bf16.h>
#include <cmath>

// ---------------------------------------------------------------------------
// FGIB forward, fused for MI455X (gfx1250, wave32, WMMA).
//  - ew (E x 44 x 44 = 1.16 GB) is never materialized: the edge einsum is
//    recomputed per iteration as 11 WMMA GEMMs per 16-edge tile with per-row
//    f32 rescale by hidden[e,d].
//  - All WMMA operands come from pre-packed / pre-padded f16 buffers so the
//    inner loops are pure b128 loads + v_wmma (no branches, no converts).
//  - k_msg_agg stages the 60 KB packed We2 panel into LDS once per workgroup;
//    B operands are then ds_load_b128 while the A gather uses the vmem pipe.
// ---------------------------------------------------------------------------

typedef __attribute__((ext_vector_type(16))) _Float16 v16h;
typedef __attribute__((ext_vector_type(8)))  _Float16 v8h;
typedef __attribute__((ext_vector_type(8)))  float    v8f;

#define NN  75000
#define NPAD 75008          // 4688 * 16 row-padded node count
#define EE  150000
#define HH  44
#define HP  48              // f32 node row stride
#define HPH 64              // f16 node row stride (K padded to 64, zero-filled)
#define EDD 10
#define NFRAG 15000
#define NGRAPH 1500
#define ROWTILES 4688       // NPAD/16
#define ETILES   9375       // EE/16
#define WE2_LDS_TILES (EDD*2*3)            // 60 tiles (d=0..9) staged in LDS
#define WE2_LDS_ELEMS (WE2_LDS_TILES*512)  // 30720 f16 = 61440 bytes

__device__ __forceinline__ float sigmoidf_(float x){ return 1.0f/(1.0f+__expf(-x)); }

// K index of element e within a lane's v16h (16-bit A/B WMMA layout):
// elements 0..7 -> K = half*8 + e ; elements 8..15 -> K = 16 + half*8 + (e-8)
__device__ __forceinline__ int wmma_k(int e, int half){
    return e + half*8 + ((e>=8)?8:0);
}

// two aligned 16-byte f16 loads -> one v16h operand
__device__ __forceinline__ v16h ld16(const _Float16* p0, const _Float16* p1){
    v8h lo = *(const v8h*)p0;
    v8h hi = *(const v8h*)p1;
    return __builtin_shufflevector(lo, hi, 0,1,2,3,4,5,6,7,8,9,10,11,12,13,14,15);
}
// A operand for K-step k0 from a padded stride-64 f16 row
__device__ __forceinline__ v16h ldA(const _Float16* rowPtr, int k0, int half){
    return ld16(rowPtr + k0 + half*8, rowPtr + k0 + 16 + half*8);
}
// B operand from packed layout: [tile][lane][16] f16, contiguous per lane
__device__ __forceinline__ v16h ldB(const _Float16* Bp, int tile, int lane){
    const _Float16* bp = Bp + ((size_t)tile*32 + lane)*16;
    return ld16(bp, bp + 8);
}

// --------------------------- packing kernels -------------------------------
// x (N x 44 f32) -> xh (NPAD x 64 f16, zero padded)
__global__ void k_packA(const float* __restrict__ x, _Float16* __restrict__ xh){
    int i = blockIdx.x*blockDim.x + threadIdx.x;
    if (i >= NPAD*HPH) return;
    int row = i / HPH, col = i % HPH;
    float v = (row < NN && col < HH) ? x[row*HH + col] : 0.0f;
    xh[i] = (_Float16)v;
}

// W (Krows x 44 f32, row-major) -> packed B tiles [kTiles*3][32][16] f16
__global__ void k_packB(const float* __restrict__ W, int Krows, int kTiles,
                        _Float16* __restrict__ out){
    int i = blockIdx.x*blockDim.x + threadIdx.x;
    int n = kTiles*3*32*16;
    if (i >= n) return;
    int e    = i & 15;
    int lane = (i >> 4) & 31;
    int ct   = (i >> 9) % 3;
    int kk   = i / (3*512);
    int half = lane >> 4, l16 = lane & 15;
    int k   = kk*32 + wmma_k(e, half);
    int col = ct*16 + l16;
    float v = (k < Krows && col < HH) ? W[k*HH + col] : 0.0f;
    out[i] = (_Float16)v;
}

// We2 (10 x 44 x 44) + be2 (44 x 44) -> packed [11][2][3][32][16]
__global__ void k_packWe2(const float* __restrict__ We2, const float* __restrict__ be2,
                          _Float16* __restrict__ out){
    int i = blockIdx.x*blockDim.x + threadIdx.x;
    const int n = 11*2*3*512;
    if (i >= n) return;
    int e    = i & 15;
    int lane = (i >> 4) & 31;
    int ct   = (i >> 9) % 3;
    int kk   = (i / (3*512)) % 2;
    int d    = i / (2*3*512);
    int half = lane >> 4, l16 = lane & 15;
    int k   = kk*32 + wmma_k(e, half);
    int col = ct*16 + l16;
    const float* Wd = (d < EDD) ? (We2 + (size_t)d*HH*HH) : be2;
    float v = (k < HH && col < HH) ? Wd[k*HH + col] : 0.0f;
    out[i] = (_Float16)v;
}

// --------------------------- edge hidden -----------------------------------
__global__ void k_edge_hidden(const float* __restrict__ ea,
                              const float* __restrict__ We1,
                              const float* __restrict__ be1,
                              float* __restrict__ hid){
    int e = blockIdx.x*blockDim.x + threadIdx.x;
    if (e >= EE) return;
    float a[EDD];
    #pragma unroll
    for (int i=0;i<EDD;i++) a[i] = ea[e*EDD+i];
    #pragma unroll
    for (int o=0;o<EDD;o++){
        float acc = be1[o];
        #pragma unroll
        for (int i=0;i<EDD;i++) acc += a[i]*We1[i*EDD+o];
        hid[e*EDD+o] = fmaxf(acc, 0.0f);
    }
}

// ---------------- out0 = relu(x@W0 + b0), WMMA, packed operands ------------
__global__ void k_gemm_out0(const _Float16* __restrict__ xh,
                            const _Float16* __restrict__ W0p,
                            const float* __restrict__ bias,
                            float* __restrict__ outF, _Float16* __restrict__ outH){
    int lane = threadIdx.x & 31;
    int wave = (blockIdx.x*blockDim.x + threadIdx.x) >> 5;
    int rt = wave/3, ct = wave%3;
    if (rt >= ROWTILES) return;
    int half = lane>>4, l16 = lane&15;
    const _Float16* ap = xh + (size_t)(rt*16 + l16)*HPH;
    v8f acc = {};
    #pragma unroll
    for (int kk=0; kk<2; ++kk){
        v16h a = ldA(ap, kk*32, half);
        v16h b = ldB(W0p, kk*3 + ct, lane);
        acc = __builtin_amdgcn_wmma_f32_16x16x32_f16(false, a, false, b,
                                                     (short)0, acc, false, false);
    }
    int col = ct*16 + l16;
    if (col < HH){
        float bv = bias[col];
        #pragma unroll
        for (int j=0;j<8;j++){
            int r = rt*16 + j + half*8;
            float v = fmaxf(acc[j] + bv, 0.0f);
            outF[(size_t)r*HP  + col] = v;
            outH[(size_t)r*HPH + col] = (_Float16)v;
        }
    }
}

// ---- fused message + aggregate: agg[dst] += out[src] @ ew (recomputed) ----
// We2 panels for d=0..9 staged in LDS (60 KB); bias panel read from global.
__global__ void __launch_bounds__(256)
k_msg_agg(const _Float16* __restrict__ outH,
          const float* __restrict__ hid,
          const _Float16* __restrict__ We2p,
          const int* __restrict__ src,
          const int* __restrict__ dst,
          float* __restrict__ agg){
    __shared__ _Float16 shB[WE2_LDS_ELEMS];

    // cooperative stage: first 60 tiles of We2p -> LDS (16B chunks)
    {
        const v8h* g = (const v8h*)We2p;
        v8h*       l = (v8h*)shB;
        for (int i = threadIdx.x; i < WE2_LDS_ELEMS/8; i += 256) l[i] = g[i];
    }
    __syncthreads();

    int lane = threadIdx.x & 31;
    int wave = (blockIdx.x*blockDim.x + threadIdx.x) >> 5;
    int et = wave/3, ct = wave%3;
    if (et >= ETILES) return;
    int half = lane>>4, l16 = lane&15;
    int s = src[et*16 + l16];
    const _Float16* ap = outH + (size_t)s*HPH;
    v16h A0 = ldA(ap, 0,  half);
    v16h A1 = ldA(ap, 32, half);

    v8f msg = {};
    for (int d=0; d<EDD; ++d){             // 10 hidden dims, B from LDS
        v8f c = {};
        {
            const _Float16* bp = &shB[(((d*2+0)*3 + ct)*32 + lane)*16];
            v8h lo = *(const v8h*)bp; v8h hi = *(const v8h*)(bp + 8);
            v16h b = __builtin_shufflevector(lo, hi, 0,1,2,3,4,5,6,7,8,9,10,11,12,13,14,15);
            c = __builtin_amdgcn_wmma_f32_16x16x32_f16(false, A0, false, b,
                                                       (short)0, c, false, false);
        }
        {
            const _Float16* bp = &shB[(((d*2+1)*3 + ct)*32 + lane)*16];
            v8h lo = *(const v8h*)bp; v8h hi = *(const v8h*)(bp + 8);
            v16h b = __builtin_shufflevector(lo, hi, 0,1,2,3,4,5,6,7,8,9,10,11,12,13,14,15);
            c = __builtin_amdgcn_wmma_f32_16x16x32_f16(false, A1, false, b,
                                                       (short)0, c, false, false);
        }
        #pragma unroll
        for (int j=0;j<8;j++){
            int er = et*16 + j + half*8;
            msg[j] += hid[er*EDD + d] * c[j];
        }
    }
    {                                      // bias matrix (scale 1), B global
        v8f c = {};
        v16h b0 = ldB(We2p, (EDD*2+0)*3 + ct, lane);
        c = __builtin_amdgcn_wmma_f32_16x16x32_f16(false, A0, false, b0,
                                                   (short)0, c, false, false);
        v16h b1 = ldB(We2p, (EDD*2+1)*3 + ct, lane);
        c = __builtin_amdgcn_wmma_f32_16x16x32_f16(false, A1, false, b1,
                                                   (short)0, c, false, false);
        #pragma unroll
        for (int j=0;j<8;j++) msg[j] += c[j];
    }
    int col = ct*16 + l16;
    if (col < HH){
        #pragma unroll
        for (int j=0;j<8;j++){
            int er = et*16 + j + half*8;
            atomicAdd(&agg[(size_t)dst[er]*HP + col], msg[j]);
        }
    }
}

// ---------------- m = relu(agg + out@root + conv_b) ------------------------
__global__ void k_root_m(const _Float16* __restrict__ outH,
                         const float* __restrict__ agg,
                         const _Float16* __restrict__ rootp,
                         const float* __restrict__ convb,
                         _Float16* __restrict__ mH){
    int lane = threadIdx.x & 31;
    int wave = (blockIdx.x*blockDim.x + threadIdx.x) >> 5;
    int rt = wave/3, ct = wave%3;
    if (rt >= ROWTILES) return;
    int half = lane>>4, l16 = lane&15;
    const _Float16* ap = outH + (size_t)(rt*16 + l16)*HPH;
    v8f acc = {};
    #pragma unroll
    for (int kk=0; kk<2; ++kk){
        v16h a = ldA(ap, kk*32, half);
        v16h b = ldB(rootp, kk*3 + ct, lane);
        acc = __builtin_amdgcn_wmma_f32_16x16x32_f16(false, a, false, b,
                                                     (short)0, acc, false, false);
    }
    int col = ct*16 + l16;
    if (col < HH){
        float cb = convb[col];
        #pragma unroll
        for (int j=0;j<8;j++){
            int r = rt*16 + j + half*8;
            float v = fmaxf(acc[j] + agg[(size_t)r*HP + col] + cb, 0.0f);
            mH[(size_t)r*HPH + col] = (_Float16)v;
        }
    }
}

// -------- out_new = m @ Wm_top + out @ Wm_bot + bm  (concat split) ---------
__global__ void k_concat_gemm(const _Float16* __restrict__ mH,
                              const _Float16* __restrict__ outH,
                              const _Float16* __restrict__ WmTp,
                              const _Float16* __restrict__ WmBp,
                              const float* __restrict__ bm,
                              float* __restrict__ noutF, _Float16* __restrict__ noutH){
    int lane = threadIdx.x & 31;
    int wave = (blockIdx.x*blockDim.x + threadIdx.x) >> 5;
    int rt = wave/3, ct = wave%3;
    if (rt >= ROWTILES) return;
    int half = lane>>4, l16 = lane&15;
    size_t rowOff = (size_t)(rt*16 + l16)*HPH;
    const _Float16* apM = mH   + rowOff;
    const _Float16* apO = outH + rowOff;
    v8f acc = {};
    #pragma unroll
    for (int kk=0; kk<2; ++kk){
        v16h a = ldA(apM, kk*32, half);
        v16h b = ldB(WmTp, kk*3 + ct, lane);
        acc = __builtin_amdgcn_wmma_f32_16x16x32_f16(false, a, false, b,
                                                     (short)0, acc, false, false);
    }
    #pragma unroll
    for (int kk=0; kk<2; ++kk){
        v16h a = ldA(apO, kk*32, half);
        v16h b = ldB(WmBp, kk*3 + ct, lane);
        acc = __builtin_amdgcn_wmma_f32_16x16x32_f16(false, a, false, b,
                                                     (short)0, acc, false, false);
    }
    int col = ct*16 + l16;
    if (col < HH){
        float bv = bm[col];
        #pragma unroll
        for (int j=0;j<8;j++){
            int r = rt*16 + j + half*8;
            float v = acc[j] + bv;            // no relu
            noutF[(size_t)r*HP  + col] = v;
            noutH[(size_t)r*HPH + col] = (_Float16)v;
        }
    }
}

// ------------- nodef = normalize(out + x); fragment accumulate -------------
__global__ void k_nodef(const float* __restrict__ outF, const float* __restrict__ x,
                        const int* __restrict__ n2f,
                        float* __restrict__ fragsum, float* __restrict__ fragcnt){
    int i = blockIdx.x*blockDim.x + threadIdx.x;
    if (i >= NN) return;
    float v[HH]; float s = 0.0f;
    #pragma unroll
    for (int h=0;h<HH;h++){ float t = outF[(size_t)i*HP+h] + x[i*HH+h]; v[h]=t; s += t*t; }
    float inv = 1.0f / fmaxf(sqrtf(s), 1e-12f);
    int f = n2f[i];
    #pragma unroll
    for (int h=0;h<HH;h++) atomicAdd(&fragsum[f*HP+h], v[h]*inv);
    atomicAdd(&fragcnt[f], 1.0f);
}

__global__ void k_fragmean(const float* __restrict__ fs, const float* __restrict__ fc,
                           float* __restrict__ frag){
    int i = blockIdx.x*blockDim.x + threadIdx.x;
    if (i >= NFRAG*HH) return;
    int f = i/HH, h = i%HH;
    frag[f*HP+h] = fs[f*HP+h] / fmaxf(fc[f], 1.0f);
}

// ---------------------------- z = frag@Wc1+bc1 -----------------------------
__global__ void k_z(const float* __restrict__ frag, const float* __restrict__ Wc1,
                    const float* __restrict__ bc1, float* __restrict__ zb){
    int i = blockIdx.x*blockDim.x + threadIdx.x;
    if (i >= NFRAG*HH) return;
    int f = i/HH, o = i%HH;
    float acc = bc1[o];
    #pragma unroll
    for (int h=0;h<HH;h++) acc += frag[f*HP+h]*Wc1[h*HH+o];
    zb[f*HP+o] = acc;
}

// -------------------- per-column mean/var over NFRAG rows ------------------
__global__ void k_colstats(const float* __restrict__ z, float* __restrict__ cmean,
                           float* __restrict__ cvar){
    __shared__ float s1[256], s2[256];
    int c = blockIdx.x, t = threadIdx.x;
    float a=0.0f, b=0.0f;
    for (int f=t; f<NFRAG; f+=256){ float v = z[f*HP+c]; a+=v; b+=v*v; }
    s1[t]=a; s2[t]=b; __syncthreads();
    for (int st=128; st>0; st>>=1){
        if (t<st){ s1[t]+=s1[t+st]; s2[t]+=s2[t+st]; }
        __syncthreads();
    }
    if (t==0){ float m = s1[0]/(float)NFRAG; cmean[c]=m; cvar[c]=s2[0]/(float)NFRAG - m*m; }
}

// ------------- w = sigmoid(relu(bn(z))@Wc2+bc2); preserve count ------------
__global__ void k_wfrag(const float* __restrict__ z, const float* __restrict__ cmean,
                        const float* __restrict__ cvar, const float* __restrict__ bng,
                        const float* __restrict__ bnb, const float* __restrict__ Wc2,
                        const float* __restrict__ bc2,
                        float* __restrict__ wbuf, float* __restrict__ scal){
    int f = blockIdx.x*blockDim.x + threadIdx.x;
    if (f >= NFRAG) return;
    float acc = bc2[0];
    #pragma unroll
    for (int h=0;h<HH;h++){
        float zn = (z[f*HP+h]-cmean[h]) * rsqrtf(cvar[h]+1e-5f) * bng[h] + bnb[h];
        acc += fmaxf(zn, 0.0f) * Wc2[h];
    }
    float w = sigmoidf_(acc);
    wbuf[f] = w;
    atomicAdd(&scal[2], (w > 0.5f) ? 1.0f : 0.0f);
}

// ----------------------- graph stats over fragments ------------------------
__global__ void k_gacc(const float* __restrict__ frag, const int* __restrict__ f2g,
                       float* __restrict__ gsum, float* __restrict__ gsq,
                       float* __restrict__ gcnt){
    int f = blockIdx.x*blockDim.x + threadIdx.x;
    if (f >= NFRAG) return;
    int g = f2g[f];
    #pragma unroll
    for (int h=0;h<HH;h++){
        float v = frag[f*HP+h];
        atomicAdd(&gsum[g*HP+h], v);
        atomicAdd(&gsq [g*HP+h], v*v);
    }
    atomicAdd(&gcnt[g], 1.0f);
}

__global__ void k_gfin(const float* __restrict__ gsum, const float* __restrict__ gsq,
                       const float* __restrict__ gcnt,
                       float* __restrict__ gmean, float* __restrict__ gstdb){
    int i = blockIdx.x*blockDim.x + threadIdx.x;
    if (i >= NGRAPH*HH) return;
    int g = i/HH, h = i%HH;
    float c  = gcnt[g];
    float gm = gsum[g*HP+h] / fmaxf(c, 1.0f);
    float gv = (gsq[g*HP+h] - c*gm*gm) / fmaxf(c-1.0f, 1.0f);
    gmean[g*HP+h] = gm;
    gstdb[g*HP+h] = sqrtf(fmaxf(gv, 0.0f));
}

// --------- noisy fragment features -> subgraph sum + KL scalars ------------
__global__ void k_noisy(const float* __restrict__ frag, const float* __restrict__ wbuf,
                        const int* __restrict__ f2g, const float* __restrict__ gmean,
                        const float* __restrict__ gstdb, const float* __restrict__ noise,
                        float* __restrict__ subsum, float* __restrict__ scal){
    int f = blockIdx.x*blockDim.x + threadIdx.x;
    if (f >= NFRAG) return;
    int g = f2g[f];
    float wv = wbuf[f], ln = 1.0f - wv;
    float t1 = 0.0f, t2 = 0.0f;
    #pragma unroll
    for (int h=0;h<HH;h++){
        float fm = frag[f*HP+h];
        float mf = gmean[g*HP+h];
        float sf = gstdb[g*HP+h];
        float nm = wv*fm + ln*mf;
        float ns = ln*sf;
        float ny = nm + noise[f*HH+h]*ns;
        atomicAdd(&subsum[g*HP+h], ny);
        float den = sf + 1e-7f;
        t1 += (ns*ns)/(den*den);
        float dd = (nm - mf)/den;
        t2 += dd*dd;
    }
    atomicAdd(&scal[0], 0.5f*t1/(float)HH);   // -> sum_g klA
    atomicAdd(&scal[1], t2);                  // -> sum_{g,h} klB
}

__global__ void k_submean(const float* __restrict__ subsum, const float* __restrict__ gcnt,
                          float* __restrict__ subb){
    int i = blockIdx.x*blockDim.x + threadIdx.x;
    if (i >= NGRAPH*HH) return;
    int g = i/HH, h = i%HH;
    subb[g*HP+h] = subsum[g*HP+h] / fmaxf(gcnt[g], 1.0f);
}

// ------------------- 44->256->128->1 MLP, one block per graph --------------
__global__ void k_pred(const float* __restrict__ sub,
                       const float* __restrict__ Wp1, const float* __restrict__ bp1,
                       const float* __restrict__ Wp2, const float* __restrict__ bp2,
                       const float* __restrict__ Wp3, const float* __restrict__ bp3,
                       float* __restrict__ pred){
    __shared__ float ss[HH];
    __shared__ float h1[256];
    __shared__ float h2[128];
    __shared__ float red[256];
    int g = blockIdx.x, t = threadIdx.x;
    if (t < HH) ss[t] = sub[g*HP + t];
    __syncthreads();
    float a = bp1[t];
    #pragma unroll
    for (int h=0;h<HH;h++) a += ss[h]*Wp1[h*256 + t];
    h1[t] = fmaxf(a, 0.0f);
    __syncthreads();
    if (t < 128){
        float b = bp2[t];
        for (int k=0;k<256;k++) b += h1[k]*Wp2[k*128 + t];
        h2[t] = fmaxf(b, 0.0f);
    }
    __syncthreads();
    red[t] = (t < 128) ? h2[t]*Wp3[t] : 0.0f;
    __syncthreads();
    for (int st=128; st>0; st>>=1){
        if (t<st) red[t] += red[t+st];
        __syncthreads();
    }
    if (t==0) pred[g] = sigmoidf_(red[0] + bp3[0]);
}

__global__ void k_final(const float* __restrict__ scal, float* __restrict__ out){
    out[NGRAPH]   = scal[0]/(float)NGRAPH + scal[1]/((float)NGRAPH*(float)HH);
    out[NGRAPH+1] = scal[2]/(float)NFRAG;
}

// ---------------------------------------------------------------------------
extern "C" void kernel_launch(void* const* d_in, const int* in_sizes, int n_in,
                              void* d_out, int out_size, void* d_ws, size_t ws_size,
                              hipStream_t stream)
{
    (void)in_sizes; (void)n_in; (void)out_size; (void)ws_size;

    const float* x    = (const float*)d_in[0];
    const float* ea   = (const float*)d_in[1];
    const float* noise= (const float*)d_in[2];
    const float* W0   = (const float*)d_in[3];
    const float* b0   = (const float*)d_in[4];
    const float* We1  = (const float*)d_in[5];
    const float* be1  = (const float*)d_in[6];
    const float* We2  = (const float*)d_in[7];
    const float* be2  = (const float*)d_in[8];
    const float* root = (const float*)d_in[9];
    const float* convb= (const float*)d_in[10];
    const float* Wm   = (const float*)d_in[11];
    const float* bm   = (const float*)d_in[12];
    const float* Wc1  = (const float*)d_in[13];
    const float* bc1  = (const float*)d_in[14];
    const float* bng  = (const float*)d_in[15];
    const float* bnb  = (const float*)d_in[16];
    const float* Wc2  = (const float*)d_in[17];
    const float* bc2  = (const float*)d_in[18];
    const float* Wp1  = (const float*)d_in[19];
    const float* bp1  = (const float*)d_in[20];
    const float* Wp2  = (const float*)d_in[21];
    const float* bp2  = (const float*)d_in[22];
    const float* Wp3  = (const float*)d_in[23];
    const float* bp3  = (const float*)d_in[24];
    const int*   eidx = (const int*)d_in[25];
    const int*   n2f  = (const int*)d_in[26];
    const int*   f2g  = (const int*)d_in[27];
    const int* src = eidx;
    const int* dst = eidx + EE;

    // ---- carve workspace (aligned 256B) ----
    char* p = (char*)d_ws;
    auto take = [&](size_t bytes)->char* {
        char* r = p; p += (bytes + 255) & ~(size_t)255; return r;
    };
    float*    hid     = (float*)   take((size_t)EE*EDD*4);
    _Float16* xh      = (_Float16*)take((size_t)NPAD*HPH*2);
    float*    outAF   = (float*)   take((size_t)NPAD*HP*4);
    _Float16* outAH   = (_Float16*)take((size_t)NPAD*HPH*2);
    float*    outBF   = (float*)   take((size_t)NPAD*HP*4);
    _Float16* outBH   = (_Float16*)take((size_t)NPAD*HPH*2);
    _Float16* mH      = (_Float16*)take((size_t)NPAD*HPH*2);
    float*    agg     = (float*)   take((size_t)NPAD*HP*4);
    _Float16* W0p     = (_Float16*)take((size_t)2*3*512*2);
    _Float16* rootp   = (_Float16*)take((size_t)2*3*512*2);
    _Float16* WmTp    = (_Float16*)take((size_t)2*3*512*2);
    _Float16* WmBp    = (_Float16*)take((size_t)2*3*512*2);
    _Float16* We2p    = (_Float16*)take((size_t)11*2*3*512*2);
    float*    fragsum = (float*)   take((size_t)NFRAG*HP*4);
    float*    fragcnt = (float*)   take((size_t)NFRAG*4);
    float*    frag    = (float*)   take((size_t)NFRAG*HP*4);
    float*    zb      = (float*)   take((size_t)NFRAG*HP*4);
    float*    cmean   = (float*)   take(HP*4);
    float*    cvar    = (float*)   take(HP*4);
    float*    wbuf    = (float*)   take((size_t)NFRAG*4);
    float*    gsum    = (float*)   take((size_t)NGRAPH*HP*4);
    float*    gsq     = (float*)   take((size_t)NGRAPH*HP*4);
    float*    gcnt    = (float*)   take((size_t)NGRAPH*4);
    float*    gmean   = (float*)   take((size_t)NGRAPH*HP*4);
    float*    gstdb   = (float*)   take((size_t)NGRAPH*HP*4);
    float*    subsum  = (float*)   take((size_t)NGRAPH*HP*4);
    float*    subb    = (float*)   take((size_t)NGRAPH*HP*4);
    float*    scal    = (float*)   take(4*4);

    // ---- zero pads / accumulators (graph-capture-safe async memsets) ----
    hipMemsetAsync(outAH,   0, (size_t)NPAD*HPH*2,   stream);  // K-pad cols must be 0
    hipMemsetAsync(outBH,   0, (size_t)NPAD*HPH*2,   stream);
    hipMemsetAsync(mH,      0, (size_t)NPAD*HPH*2,   stream);
    hipMemsetAsync(fragsum, 0, (size_t)NFRAG*HP*4,   stream);
    hipMemsetAsync(fragcnt, 0, (size_t)NFRAG*4,      stream);
    hipMemsetAsync(gsum,    0, (size_t)NGRAPH*HP*4,  stream);
    hipMemsetAsync(gsq,     0, (size_t)NGRAPH*HP*4,  stream);
    hipMemsetAsync(gcnt,    0, (size_t)NGRAPH*4,     stream);
    hipMemsetAsync(subsum,  0, (size_t)NGRAPH*HP*4,  stream);
    hipMemsetAsync(scal,    0, 16,                   stream);

    // ---- pack activations & weights into WMMA-friendly f16 layouts ----
    k_packA  <<<(NPAD*HPH+255)/256, 256, 0, stream>>>(x, xh);
    k_packB  <<<(2*3*512+255)/256,  256, 0, stream>>>(W0,   HH,   2, W0p);
    k_packB  <<<(2*3*512+255)/256,  256, 0, stream>>>(root, HH,   2, rootp);
    k_packB  <<<(2*3*512+255)/256,  256, 0, stream>>>(Wm,          HH, 2, WmTp);
    k_packB  <<<(2*3*512+255)/256,  256, 0, stream>>>(Wm + HH*HH,  HH, 2, WmBp);
    k_packWe2<<<(11*2*3*512+255)/256, 256, 0, stream>>>(We2, be2, We2p);

    // ---- edge hidden (E x 10) ----
    k_edge_hidden<<<(EE+255)/256, 256, 0, stream>>>(ea, We1, be1, hid);

    // ---- out0 = relu(x @ W0 + b0) via WMMA ----
    const int gemmBlocks = (ROWTILES*3*32 + 255)/256;
    k_gemm_out0<<<gemmBlocks, 256, 0, stream>>>(xh, W0p, b0, outAF, outAH);

    // ---- 3 fused message-passing iterations ----
    const int msgBlocks = (ETILES*3*32 + 255)/256;
    float* curF = outAF; _Float16* curH = outAH;
    float* nxtF = outBF; _Float16* nxtH = outBH;
    for (int it=0; it<3; ++it){
        hipMemsetAsync(agg, 0, (size_t)NPAD*HP*4, stream);
        k_msg_agg    <<<msgBlocks,  256, 0, stream>>>(curH, hid, We2p, src, dst, agg);
        k_root_m     <<<gemmBlocks, 256, 0, stream>>>(curH, agg, rootp, convb, mH);
        k_concat_gemm<<<gemmBlocks, 256, 0, stream>>>(mH, curH, WmTp, WmBp, bm, nxtF, nxtH);
        float* tf = curF; curF = nxtF; nxtF = tf;
        _Float16* th = curH; curH = nxtH; nxtH = th;
    }

    // ---- node features -> fragment pooling -> weights -> graph stats ----
    k_nodef   <<<(NN+255)/256,        256, 0, stream>>>(curF, x, n2f, fragsum, fragcnt);
    k_fragmean<<<(NFRAG*HH+255)/256,  256, 0, stream>>>(fragsum, fragcnt, frag);
    k_z       <<<(NFRAG*HH+255)/256,  256, 0, stream>>>(frag, Wc1, bc1, zb);
    k_colstats<<<HH,                  256, 0, stream>>>(zb, cmean, cvar);
    k_wfrag   <<<(NFRAG+255)/256,     256, 0, stream>>>(zb, cmean, cvar, bng, bnb,
                                                        Wc2, bc2, wbuf, scal);
    k_gacc    <<<(NFRAG+255)/256,     256, 0, stream>>>(frag, f2g, gsum, gsq, gcnt);
    k_gfin    <<<(NGRAPH*HH+255)/256, 256, 0, stream>>>(gsum, gsq, gcnt, gmean, gstdb);
    k_noisy   <<<(NFRAG+255)/256,     256, 0, stream>>>(frag, wbuf, f2g, gmean, gstdb,
                                                        noise, subsum, scal);
    k_submean <<<(NGRAPH*HH+255)/256, 256, 0, stream>>>(subsum, gcnt, subb);

    // ---- prediction MLP + scalar outputs ----
    k_pred <<<NGRAPH, 256, 0, stream>>>(subb, Wp1, bp1, Wp2, bp2, Wp3, bp3, (float*)d_out);
    k_final<<<1, 1, 0, stream>>>(scal, (float*)d_out);
}
